// DecoderStage_22617297781337
// MI455X (gfx1250) — matrix operand order repaired and checked by reading, not compiled, not run on hardware
//
#include <hip/hip_runtime.h>
#include <stdint.h>

// ---- problem constants (match reference) ----
#define N_IN    20000
#define C_IN    256
#define K_UP    16
#define C_OUT   128
#define N_SKIP  160000
#define C_SKIP  64
#define N_OUTP  (N_IN * K_UP)        // 320000
#define K_FUSE  27
#define C_FEAT  (C_OUT + C_SKIP)     // 192
#define BN_EPS  1e-5f
#define BROWS   64                   // stage-B output rows per block

// ---- WMMA types ----
typedef __bf16  v16bf __attribute__((ext_vector_type(16)));
typedef float   v8f   __attribute__((ext_vector_type(8)));

union FragBF { uint4 u[2]; v16bf v; };

__device__ __forceinline__ uint16_t f2bf(float f) {
    uint32_t u = __float_as_uint(f);
    uint32_t r = u + 0x7FFFu + ((u >> 16) & 1u);   // round-to-nearest-even
    return (uint16_t)(r >> 16);
}

// -------------------------------------------------------------------------
// 1) fp32 -> bf16 convert of x_feats (pairs)
// -------------------------------------------------------------------------
__global__ __launch_bounds__(256) void k_convert_x(const float* __restrict__ x,
                                                   uint16_t* __restrict__ xb) {
    int id = blockIdx.x * 256 + threadIdx.x;               // one float2 each
    const int total = N_IN * C_IN / 2;
    if (id >= total) return;
    float2 v = ((const float2*)x)[id];
    uint32_t p = (uint32_t)f2bf(v.x) | ((uint32_t)f2bf(v.y) << 16);
    ((uint32_t*)xb)[id] = p;
}

// -------------------------------------------------------------------------
// 2) pack w_up [16,256,128] -> B-fragment layout
//    wup[(kcb*2048 + n)*32 + half*16 + kk] = w_up[k= n>>7][c= kcb*32+half*16+kk][d= n&127]
// -------------------------------------------------------------------------
__global__ __launch_bounds__(256) void k_pack_wup(const float* __restrict__ w,
                                                  uint16_t* __restrict__ wup) {
    int o = blockIdx.x * 256 + threadIdx.x;                // 8*2048*2*16 = 524288
    if (o >= 8 * 2048 * 32) return;
    int kk   = o & 15;
    int half = (o >> 4) & 1;
    int n    = (o >> 5) & 2047;
    int kcb  = o >> 16;
    int c = kcb * 32 + half * 16 + kk;
    int k = n >> 7, d = n & 127;
    wup[o] = f2bf(w[((size_t)k * C_IN + c) * C_OUT + d]);
}

// -------------------------------------------------------------------------
// 3) pack w_fuse [27,192,128] -> B-fragment layout
//    wfu[(((k*6+kcb)*128 + n)*2 + half)*16 + kk] = w_fuse[k][c=kcb*32+half*16+kk][n]
// -------------------------------------------------------------------------
__global__ __launch_bounds__(256) void k_pack_wfuse(const float* __restrict__ w,
                                                    uint16_t* __restrict__ wfu) {
    int o = blockIdx.x * 256 + threadIdx.x;                // 27*6*128*32 = 663552
    if (o >= K_FUSE * 6 * 128 * 32) return;
    int kk   = o & 15;
    int half = (o >> 4) & 1;
    int n    = (o >> 5) & 127;
    int rest = o >> 12;                                    // kcb + 6*k
    int kcb  = rest % 6;
    int k    = rest / 6;
    int c = kcb * 32 + half * 16 + kk;
    wfu[o] = f2bf(w[((size_t)k * C_FEAT + c) * C_OUT + n]);
}

// -------------------------------------------------------------------------
// 4) skip gather: feat[:,128:192] = bf16(skip_feats[skip_idx])
// -------------------------------------------------------------------------
__global__ __launch_bounds__(256) void k_skip_gather(const float* __restrict__ skip,
                                                     const int* __restrict__ sidx,
                                                     uint16_t* __restrict__ featb) {
    int id = blockIdx.x * 256 + threadIdx.x;               // 32 threads per row
    const int total = N_OUTP * 32;
    if (id >= total) return;
    int row = id >> 5;
    int l   = id & 31;                                     // covers 64 cols, 2 each
    int si  = sidx[row];
    float2 v = ((const float2*)(skip + (size_t)si * C_SKIP))[l];
    uint32_t p = (uint32_t)f2bf(v.x) | ((uint32_t)f2bf(v.y) << 16);
    *(uint32_t*)(featb + (size_t)row * C_FEAT + C_OUT + l * 2) = p;
}

// -------------------------------------------------------------------------
// 5) Stage A: up = x @ Wup, BN1+ReLU, reshape-scatter to feat[:,0:128] (bf16)
//    grid = (1250 M-tiles, 16 = K_UP), block = 256 (8 waves, 16 cols each)
// -------------------------------------------------------------------------
__global__ __launch_bounds__(256) void k_stage_a(const uint16_t* __restrict__ xb,
                                                 const uint16_t* __restrict__ wup,
                                                 const float* __restrict__ g1,
                                                 const float* __restrict__ b1,
                                                 const float* __restrict__ m1,
                                                 const float* __restrict__ v1,
                                                 uint16_t* __restrict__ featb) {
    __shared__ __align__(16) uint16_t ldsA[16 * C_IN];     // 8 KB

    const int t    = threadIdx.x;
    const int row  = t >> 4;
    const int c16  = t & 15;
    // stage A tile 16x256 bf16 into LDS (contiguous)
    {
        const uint2* src = (const uint2*)(xb + (size_t)(blockIdx.x * 16 + row) * C_IN);
        uint2* dst = (uint2*)(ldsA + row * C_IN);
        #pragma unroll
        for (int j = 0; j < 4; ++j) dst[c16 + j * 16] = src[c16 + j * 16];
    }
    __syncthreads();

    const int wave = t >> 5;
    const int lane = t & 31;
    const int hi   = lane >> 4;
    const int m    = lane & 15;
    const int n    = blockIdx.y * 128 + wave * 16 + m;     // column in [0,2048)

    v8f acc = {0.f, 0.f, 0.f, 0.f, 0.f, 0.f, 0.f, 0.f};
    #pragma unroll
    for (int kcb = 0; kcb < 8; ++kcb) {
        FragBF a, b;
        const uint4* pa = (const uint4*)(ldsA + m * C_IN + kcb * 32 + hi * 16);
        a.u[0] = pa[0];                                    // K = 32kcb + 16hi + 0..7
        a.u[1] = pa[1];                                    // K = 32kcb + 16hi + 8..15
        const uint4* pb = (const uint4*)(wup + (((size_t)kcb * 2048 + n) * 2 + hi) * 16);
        b.u[0] = pb[0];
        b.u[1] = pb[1];
        acc = __builtin_amdgcn_wmma_f32_16x16x32_bf16(false, a.v, false, b.v,
                                                      (short)0, acc, false, false);
    }

    // epilogue: BN1 + ReLU, scatter to feat row = point*16 + k_up (k_up == blockIdx.y)
    const int cout = wave * 16 + m;                        // 0..127
    const float inv  = g1[cout] * rsqrtf(v1[cout] + BN_EPS);
    const float bias = b1[cout] - m1[cout] * inv;
    #pragma unroll
    for (int i = 0; i < 8; ++i) {
        int pt = blockIdx.x * 16 + i + hi * 8;
        float val = fmaxf(acc[i] * inv + bias, 0.f);
        featb[(size_t)(pt * K_UP + blockIdx.y) * C_FEAT + cout] = f2bf(val);
    }
}

// -------------------------------------------------------------------------
// 6) Stage B: out = BN2ReLU( sum_k mask * feat[nbr[:,k]] @ Wfuse[k] )
//    grid = 5000 (64-row tiles), block = 256 (8 waves x 16 cols = 128 cols)
//    - 64-row blocking: each B fragment feeds 4 wmma (4x less w_fuse L2 traffic)
//    - double-buffered LDS gather: one barrier per k, gather k+1 overlaps math k
// -------------------------------------------------------------------------
__global__ __launch_bounds__(256) void k_stage_b(const uint16_t* __restrict__ featb,
                                                 const uint16_t* __restrict__ wfu,
                                                 const int* __restrict__ nbr_idx,
                                                 const uint8_t* __restrict__ nbr_mask,
                                                 const float* __restrict__ g2,
                                                 const float* __restrict__ b2,
                                                 const float* __restrict__ m2,
                                                 const float* __restrict__ v2,
                                                 float* __restrict__ out) {
    __shared__ __align__(16) uint16_t ldsA[2][BROWS * C_FEAT];   // 2 x 24 KB

    const int t    = threadIdx.x;
    const int srow = t >> 2;                               // staging: 4 thr / row
    const int q    = t & 3;
    const int wave = t >> 5;
    const int lane = t & 31;
    const int hi   = lane >> 4;
    const int m    = lane & 15;
    const int rbase = blockIdx.x * BROWS;

    v8f acc[4];
    #pragma unroll
    for (int mt = 0; mt < 4; ++mt)
        acc[mt] = (v8f){0.f, 0.f, 0.f, 0.f, 0.f, 0.f, 0.f, 0.f};

    // masked gather of one 64-row feature tile (192 bf16/row) into LDS buffer
    auto stage = [&](int k, int buf) {
        int     idx = nbr_idx[(size_t)(rbase + srow) * K_FUSE + k];
        uint8_t msk = nbr_mask[(size_t)(rbase + srow) * K_FUSE + k];
        uint4* dst = (uint4*)(ldsA[buf] + srow * C_FEAT);  // 24 uint4 per row
        if (msk) {
            const uint4* src = (const uint4*)(featb + (size_t)idx * C_FEAT);
            #pragma unroll
            for (int j = 0; j < 6; ++j) dst[q + j * 4] = src[q + j * 4];
        } else {
            uint4 z; z.x = 0u; z.y = 0u; z.z = 0u; z.w = 0u;
            #pragma unroll
            for (int j = 0; j < 6; ++j) dst[q + j * 4] = z;
        }
    };

    stage(0, 0);
    for (int k = 0; k < K_FUSE; ++k) {
        __syncthreads();            // staging(k) done AND compute(k-1) done
        if (k + 1 < K_FUSE) stage(k + 1, (k + 1) & 1);
        const uint16_t* lds = ldsA[k & 1];

        #pragma unroll
        for (int kcb = 0; kcb < 6; ++kcb) {
            FragBF b;
            const uint4* pb = (const uint4*)
                (wfu + (((size_t)(k * 6 + kcb) * 128 + wave * 16 + m) * 2 + hi) * 16);
            b.u[0] = pb[0];
            b.u[1] = pb[1];
            #pragma unroll
            for (int mt = 0; mt < 4; ++mt) {
                FragBF a;
                const uint4* pa = (const uint4*)
                    (lds + (mt * 16 + m) * C_FEAT + kcb * 32 + hi * 16);
                a.u[0] = pa[0];                            // K = 32kcb + 16hi + 0..7
                a.u[1] = pa[1];                            // K = 32kcb + 16hi + 8..15
                acc[mt] = __builtin_amdgcn_wmma_f32_16x16x32_bf16(
                    false, a.v, false, b.v, (short)0, acc[mt], false, false);
            }
        }
    }

    // epilogue: BN2 + ReLU -> fp32 out
    const int col = wave * 16 + m;
    const float inv  = g2[col] * rsqrtf(v2[col] + BN_EPS);
    const float bias = b2[col] - m2[col] * inv;
    #pragma unroll
    for (int mt = 0; mt < 4; ++mt) {
        #pragma unroll
        for (int i = 0; i < 8; ++i) {
            int r = rbase + mt * 16 + i + hi * 8;
            out[(size_t)r * C_OUT + col] = fmaxf(acc[mt][i] * inv + bias, 0.f);
        }
    }
}

// -------------------------------------------------------------------------
extern "C" void kernel_launch(void* const* d_in, const int* in_sizes, int n_in,
                              void* d_out, int out_size, void* d_ws, size_t ws_size,
                              hipStream_t stream) {
    const float* x_feats  = (const float*)d_in[0];
    const float* w_up     = (const float*)d_in[1];
    const float* bn1_g    = (const float*)d_in[2];
    const float* bn1_b    = (const float*)d_in[3];
    const float* bn1_m    = (const float*)d_in[4];
    const float* bn1_v    = (const float*)d_in[5];
    const float* skip     = (const float*)d_in[6];
    const float* w_fuse   = (const float*)d_in[7];
    const float* bn2_g    = (const float*)d_in[8];
    const float* bn2_b    = (const float*)d_in[9];
    const float* bn2_m    = (const float*)d_in[10];
    const float* bn2_v    = (const float*)d_in[11];
    const int*   skip_idx = (const int*)d_in[12];
    const int*   nbr_idx  = (const int*)d_in[13];
    const uint8_t* nbr_mask = (const uint8_t*)d_in[14];
    float* out = (float*)d_out;

    // workspace layout (bf16 buffers as uint16_t)
    uint16_t* xb    = (uint16_t*)d_ws;                      // 20000*256
    uint16_t* wup   = xb  + (size_t)N_IN * C_IN;            // 524288
    uint16_t* wfu   = wup + (size_t)8 * 2048 * 32;          // 663552
    uint16_t* featb = wfu + (size_t)K_FUSE * 6 * 128 * 32;  // 320000*192

    k_convert_x <<<(N_IN * C_IN / 2 + 255) / 256, 256, 0, stream>>>(x_feats, xb);
    k_pack_wup  <<<(8 * 2048 * 32 + 255) / 256, 256, 0, stream>>>(w_up, wup);
    k_pack_wfuse<<<(K_FUSE * 6 * 128 * 32 + 255) / 256, 256, 0, stream>>>(w_fuse, wfu);
    k_skip_gather<<<(N_OUTP * 32 + 255) / 256, 256, 0, stream>>>(skip, skip_idx, featb);

    dim3 gA(N_IN / 16, K_UP);                               // 1250 x 16
    k_stage_a<<<gA, 256, 0, stream>>>(xb, wup, bn1_g, bn1_b, bn1_m, bn1_v, featb);

    k_stage_b<<<N_OUTP / BROWS, 256, 0, stream>>>(featb, wfu, nbr_idx, nbr_mask,
                                                  bn2_g, bn2_b, bn2_m, bn2_v, out);
}
